// MPNN_9216999817552
// MI455X (gfx1250) — matrix operand order repaired and checked by reading, not compiled, not run on hardware
//
#include <hip/hip_runtime.h>

typedef __attribute__((ext_vector_type(16))) __bf16 v16bf;
typedef __attribute__((ext_vector_type(8)))  float  v8f;

#define RCUT 5.0f
#define PI_F 3.14159265358979f

__device__ __forceinline__ float sspf(float z) {
    // shifted softplus: log(1+e^z) - log(2), numerically stable
    float sp = fmaxf(z, 0.0f) + log1pf(__expf(-fabsf(z)));
    return sp - 0.69314718055994531f;
}

// ---------------------------------------------------------------------------
// Kernel 0: one-time weight prep -> transposed, padded, bf16 staging in ws
//   W1t_g [128][72]  : Wf1^T (k padded 50->64, cols 64..71 pad/zero)
//   W2t_g [128][136] : Wf2^T (cols 128..135 pad/zero)
// ---------------------------------------------------------------------------
__global__ __launch_bounds__(256) void prep_weights(
    const float* __restrict__ Wf1, const float* __restrict__ Wf2,
    __bf16* __restrict__ W1t_g, __bf16* __restrict__ W2t_g)
{
    const int stride = blockDim.x * gridDim.x;
    int tid = threadIdx.x + blockIdx.x * blockDim.x;
    for (int idx = tid; idx < 128 * 72; idx += stride) {
        int n = idx / 72, k = idx % 72;
        float v = (k < 50) ? Wf1[k * 128 + n] : 0.0f;
        W1t_g[idx] = (__bf16)v;
    }
    for (int idx = tid; idx < 128 * 136; idx += stride) {
        int n = idx / 136, k = idx % 136;
        float v = (k < 128) ? Wf2[k * 128 + n] : 0.0f;
        W2t_g[idx] = (__bf16)v;
    }
}

// ---------------------------------------------------------------------------
// Kernel 1: per-atom projections Q = x@Wq, K = x@Wk, V = x@Wv  (bf16 WMMA)
// grid.x = 4096/16 = 256 M-tiles, grid.y = 3 (q,k,v); block = 256 thr (8 waves)
// ---------------------------------------------------------------------------
__global__ __launch_bounds__(256) void qkv_kernel(
    const float* __restrict__ x,
    const float* __restrict__ Wq, const float* __restrict__ Wk,
    const float* __restrict__ Wv,
    float* __restrict__ Q, float* __restrict__ Kp, float* __restrict__ Vp)
{
    __shared__ __bf16 At[16][136];   // 16x128 A-tile (bf16), padded stride
    __shared__ __bf16 Bt[128][136];  // weight transposed [n][k]
    __shared__ float  Ot[16][132];   // f32 output staging for coalesced stores

    const int tid  = threadIdx.x;
    const int wave = tid >> 5, lane = tid & 31;
    const int half = lane >> 4, l15 = lane & 15;
    const int mbase = blockIdx.x * 16;

    const float* W = (blockIdx.y == 0) ? Wq : ((blockIdx.y == 1) ? Wk : Wv);
    float*       O = (blockIdx.y == 0) ? Q  : ((blockIdx.y == 1) ? Kp : Vp);

    for (int idx = tid; idx < 16 * 128; idx += 256) {
        int r = idx >> 7, c = idx & 127;
        At[r][c] = (__bf16)x[(mbase + r) * 128 + c];
    }
    for (int idx = tid; idx < 128 * 128; idx += 256) {
        int k = idx >> 7, n = idx & 127;   // coalesced read of W[k][n]
        Bt[n][k] = (__bf16)W[idx];
    }
    __syncthreads();

    const int nt = wave;                   // each wave owns one 16-wide N-tile
    v8f acc = {};
#pragma unroll
    for (int ks = 0; ks < 4; ++ks) {
        v16bf a, b;
        const int kb1 = ks * 32 + half * 8;
#pragma unroll
        for (int j = 0; j < 8; ++j) a[j]     = At[l15][kb1 + j];
#pragma unroll
        for (int j = 0; j < 8; ++j) a[8 + j] = At[l15][kb1 + 16 + j];
        const int brow = nt * 16 + l15;
#pragma unroll
        for (int j = 0; j < 8; ++j) b[j]     = Bt[brow][kb1 + j];
#pragma unroll
        for (int j = 0; j < 8; ++j) b[8 + j] = Bt[brow][kb1 + 16 + j];
        acc = __builtin_amdgcn_wmma_f32_16x16x32_bf16(
                  false, a, false, b, (short)0, acc, false, false);
    }
#pragma unroll
    for (int j = 0; j < 8; ++j)
        Ot[half * 8 + j][nt * 16 + l15] = acc[j];
    __syncthreads();

    // coalesced float4 stores
    for (int i = tid; i < 16 * 32; i += 256) {
        int r = i >> 5, c4 = (i & 31) << 2;
        float4 v = *(const float4*)&Ot[r][c4];
        *(float4*)&O[(mbase + r) * 128 + c4] = v;
    }
}

// ---------------------------------------------------------------------------
// Kernel 2: per-atom filter network (2 WMMA GEMMs) + cutoff + masked attention
// grid.x = B*Na = 4096 blocks; block = 256 threads (8 waves)
// ---------------------------------------------------------------------------
__global__ __launch_bounds__(256) void filter_attn_kernel(
    const float* __restrict__ r_ij, const int* __restrict__ neighbors,
    const int* __restrict__ pmask,  const float* __restrict__ f_ij,
    const __bf16* __restrict__ W1t_g, const float* __restrict__ bf1,
    const __bf16* __restrict__ W2t_g, const float* __restrict__ bf2,
    const float* __restrict__ Wo,   const float* __restrict__ bo,
    const float* __restrict__ Q,    const float* __restrict__ Kp,
    const float* __restrict__ Vp,
    float* __restrict__ m_out, float* __restrict__ W_out)
{
    constexpr int Na = 2048, Nn = 48, G = 50, F = 128;

    __shared__ __bf16 A1 [48][72];    // (2*f_ij-1), G padded 50->64
    __shared__ __bf16 W1t[128][72];   // Wf1^T  [n][k] (copied from staging)
    __shared__ __bf16 H  [48][136];   // hidden after ssp (bf16)
    __shared__ __bf16 W2t[128][136];  // Wf2^T  [n][k] (copied from staging)
    __shared__ float  Wl [48][132];   // cutoff-modulated filter (f32)
    __shared__ float  qv[128], mvec[128], sc[49], at[49], Cc[48];
    __shared__ int    nbr[48], pm[48];

    const int tid  = threadIdx.x;
    const int wave = tid >> 5, lane = tid & 31;
    const int half = lane >> 4, l15 = lane & 15;
    const int ai   = blockIdx.x;              // flat (b*Na + i)
    const int bb   = ai / Na;
    const long base_nn = (long)ai * Nn;

    __builtin_prefetch(f_ij + base_nn * G, 0, 1);   // global_prefetch_b8

    if (tid < Nn) {
        nbr[tid] = neighbors[base_nn + tid];
        pm[tid]  = pmask[base_nn + tid];
        float r  = r_ij[base_nn + tid];
        Cc[tid]  = (r < RCUT) ? 0.5f * (__cosf(PI_F * r / RCUT) + 1.0f) : 0.0f;
    }
    if (tid < F) qv[tid] = Q[(long)ai * F + tid];

    for (int idx = tid; idx < 48 * 64; idx += 256) {
        int r = idx >> 6, g = idx & 63;
        float v = (g < G) ? (2.0f * f_ij[(base_nn + r) * G + g] - 1.0f) : 0.0f;
        A1[r][g] = (__bf16)v;
    }
    // straight uint4 copies of pre-transposed bf16 weights (layout matches LDS)
    {
        const uint4* g1 = (const uint4*)W1t_g;
        uint4*       s1 = (uint4*)&W1t[0][0];
        for (int i = tid; i < (128 * 72 * 2) / 16; i += 256) s1[i] = g1[i];
        const uint4* g2 = (const uint4*)W2t_g;
        uint4*       s2 = (uint4*)&W2t[0][0];
        for (int i = tid; i < (128 * 136 * 2) / 16; i += 256) s2[i] = g2[i];
    }
    __syncthreads();

    // ---- GEMM1: [48x64] @ [64x128] -> ssp -> H (bf16) ----
#pragma unroll
    for (int mt = 0; mt < 3; ++mt) {
        const int nt = wave;
        v8f acc = {};
#pragma unroll
        for (int ks = 0; ks < 2; ++ks) {
            v16bf a, b;
            const int arow = mt * 16 + l15;
            const int kb1  = ks * 32 + half * 8;
#pragma unroll
            for (int j = 0; j < 8; ++j) a[j]     = A1[arow][kb1 + j];
#pragma unroll
            for (int j = 0; j < 8; ++j) a[8 + j] = A1[arow][kb1 + 16 + j];
            const int brow = nt * 16 + l15;
#pragma unroll
            for (int j = 0; j < 8; ++j) b[j]     = W1t[brow][kb1 + j];
#pragma unroll
            for (int j = 0; j < 8; ++j) b[8 + j] = W1t[brow][kb1 + 16 + j];
            acc = __builtin_amdgcn_wmma_f32_16x16x32_bf16(
                      false, a, false, b, (short)0, acc, false, false);
        }
#pragma unroll
        for (int j = 0; j < 8; ++j) {
            int row = mt * 16 + half * 8 + j;
            int col = nt * 16 + l15;
            H[row][col] = (__bf16)sspf(acc[j] + bf1[col]);
        }
    }
    __syncthreads();

    // ---- GEMM2: [48x128] @ [128x128] -> +bias, *cutoff -> Wl (LDS) ----
#pragma unroll
    for (int mt = 0; mt < 3; ++mt) {
        const int nt = wave;
        v8f acc = {};
#pragma unroll
        for (int ks = 0; ks < 4; ++ks) {
            v16bf a, b;
            const int arow = mt * 16 + l15;
            const int kb1  = ks * 32 + half * 8;
#pragma unroll
            for (int j = 0; j < 8; ++j) a[j]     = H[arow][kb1 + j];
#pragma unroll
            for (int j = 0; j < 8; ++j) a[8 + j] = H[arow][kb1 + 16 + j];
            const int brow = nt * 16 + l15;
#pragma unroll
            for (int j = 0; j < 8; ++j) b[j]     = W2t[brow][kb1 + j];
#pragma unroll
            for (int j = 0; j < 8; ++j) b[8 + j] = W2t[brow][kb1 + 16 + j];
            acc = __builtin_amdgcn_wmma_f32_16x16x32_bf16(
                      false, a, false, b, (short)0, acc, false, false);
        }
#pragma unroll
        for (int j = 0; j < 8; ++j) {
            int row = mt * 16 + half * 8 + j;   // neighbor slot 0..47
            int col = nt * 16 + l15;
            Wl[row][col] = (acc[j] + bf2[col]) * Cc[row];
        }
    }
    __syncthreads();

    // ---- stream W (the big HBM output) with coalesced float4 stores ----
    for (int i = tid; i < 48 * 32; i += 256) {
        int r = i >> 5, c4 = (i & 31) << 2;
        float4 v = *(const float4*)&Wl[r][c4];
        *(float4*)&W_out[(base_nn + r) * F + c4] = v;
    }

    // ---- attention scores: q . k_gathered / sqrt(F), masked ----
    for (int j = wave; j < Nn + 1; j += 8) {
        int gi = (j == 0) ? ai : (bb * Na + nbr[j - 1]);
        const float* kr = Kp + (long)gi * F;
        float p = 0.0f;
#pragma unroll
        for (int e = 0; e < 4; ++e) p += qv[lane * 4 + e] * kr[lane * 4 + e];
#pragma unroll
        for (int off = 16; off > 0; off >>= 1) p += __shfl_xor(p, off, 32);
        if (lane == 0) {
            float s = p * 0.08838834764831845f;        // 1/sqrt(128)
            if (j > 0 && pm[j - 1] == 0) s = -1e9f;
            sc[j] = s;
        }
    }
    __syncthreads();

    // ---- softmax over 49 entries ----
    if (tid == 0) {
        float mx = -1e30f;
        for (int j = 0; j < Nn + 1; ++j) mx = fmaxf(mx, sc[j]);
        float sum = 0.0f;
        for (int j = 0; j < Nn + 1; ++j) { float e = __expf(sc[j] - mx); at[j] = e; sum += e; }
        float inv = 1.0f / sum;
        for (int j = 0; j < Nn + 1; ++j) at[j] *= inv;
    }
    __syncthreads();

    // ---- aggregation: m = attn0*v_self + sum_n attn_n * (v_n * W_n) ----
    if (tid < F) {
        float acc = at[0] * Vp[(long)ai * F + tid];
        for (int n = 0; n < Nn; ++n)
            acc += at[n + 1] * Vp[((long)bb * Na + nbr[n]) * F + tid] * Wl[n][tid];
        mvec[tid] = acc;
    }
    __syncthreads();

    // ---- output projection + ssp ----
    if (tid < F) {
        float o = bo[tid];
        for (int k = 0; k < F; ++k) o += mvec[k] * Wo[k * F + tid];
        m_out[(long)ai * F + tid] = sspf(o);
    }
}

extern "C" void kernel_launch(void* const* d_in, const int* in_sizes, int n_in,
                              void* d_out, int out_size, void* d_ws, size_t ws_size,
                              hipStream_t stream) {
    // input order: e, x, t, r_ij, neighbors, pairwise_mask, f_ij,
    //              Wf1, bf1, Wf2, bf2, Wq, Wk, Wv, Wo, bo
    const float* x         = (const float*)d_in[1];
    const float* r_ij      = (const float*)d_in[3];
    const int*   neighbors = (const int*)  d_in[4];
    const int*   pmask     = (const int*)  d_in[5];
    const float* f_ij      = (const float*)d_in[6];
    const float* Wf1       = (const float*)d_in[7];
    const float* bf1       = (const float*)d_in[8];
    const float* Wf2       = (const float*)d_in[9];
    const float* bf2       = (const float*)d_in[10];
    const float* Wq        = (const float*)d_in[11];
    const float* Wk        = (const float*)d_in[12];
    const float* Wv        = (const float*)d_in[13];
    const float* Wo        = (const float*)d_in[14];
    const float* bo        = (const float*)d_in[15];

    constexpr int BNa = 2 * 2048, F = 128;

    float* m_out = (float*)d_out;                 // [B,Na,F]
    float* W_out = m_out + (long)BNa * F;         // [B,Na,Nn,F]
    (void)out_size; (void)n_in; (void)in_sizes; (void)ws_size;

    float* Q  = (float*)d_ws;                     // 3 x (4096*128) f32 = 6 MB
    float* Kp = Q  + (long)BNa * F;
    float* Vp = Kp + (long)BNa * F;
    __bf16* W1t_g = (__bf16*)(Vp + (long)BNa * F);    // [128][72]  bf16
    __bf16* W2t_g = W1t_g + 128 * 72;                 // [128][136] bf16

    prep_weights<<<32, 256, 0, stream>>>(Wf1, Wf2, W1t_g, W2t_g);

    dim3 g1(BNa / 16, 3);
    qkv_kernel<<<g1, 256, 0, stream>>>(x, Wq, Wk, Wv, Q, Kp, Vp);

    filter_attn_kernel<<<BNa, 256, 0, stream>>>(
        r_ij, neighbors, pmask, f_ij, W1t_g, bf1, W2t_g, bf2, Wo, bo,
        Q, Kp, Vp, m_out, W_out);
}
